// GCNEncoder_5892695130195
// MI455X (gfx1250) — matrix operand order repaired and checked by reading, not compiled, not run on hardware
//
#include <hip/hip_runtime.h>

typedef __attribute__((ext_vector_type(16))) __bf16 v16bf;
typedef __attribute__((ext_vector_type(8)))  float  v8f;
typedef __attribute__((ext_vector_type(8)))  __bf16 bf16x8;

#define DIM_IN 128
#define DIM_HID 128
#define DIM_OUT 64

static __device__ __forceinline__ void atomic_add_f32(float* p, float v) {
    __hip_atomic_fetch_add(p, v, __ATOMIC_RELAXED, __HIP_MEMORY_SCOPE_AGENT);
}

// ---------------- utility kernels ----------------

__global__ void k_zero_f32(float* __restrict__ p, long long count) {
    long long i = (long long)blockIdx.x * blockDim.x + threadIdx.x;
    long long stride = (long long)gridDim.x * blockDim.x;
    for (; i < count; i += stride) p[i] = 0.0f;
}

__global__ void k_cast_bf16(const float* __restrict__ src, __bf16* __restrict__ dst,
                            long long count) {
    long long i = (long long)blockIdx.x * blockDim.x + threadIdx.x;
    long long stride = (long long)gridDim.x * blockDim.x;
    for (; i < count; i += stride) dst[i] = (__bf16)src[i];
}

// deg[d] += 1 per edge (self-loop handled in k_dinv)
__global__ void k_deg_count(const int* __restrict__ dstIdx, float* __restrict__ deg, int E) {
    int e = blockIdx.x * blockDim.x + threadIdx.x;
    if (e < E) atomic_add_f32(&deg[dstIdx[e]], 1.0f);
}

__global__ void k_dinv(float* __restrict__ degInOut, int n) {
    int i = blockIdx.x * blockDim.x + threadIdx.x;
    if (i < n) degInOut[i] = rsqrtf(degInOut[i] + 1.0f);   // +1 = self loop
}

// Cast + swizzle weight matrix W[128, ncol] (fp32, row-major) into fragment-major bf16:
// Bsw[ ((ct*4 + kt)*32 + lane)*16 + q ] = W[K, ct*16 + (lane&15)],
//   K = kt*32 + (lane>>4)*16 + q      (same contiguous k = 16*hi + s order as the GEMM's A loads)
// => each lane's B fragment is 16 contiguous bf16 (32B) = two b128 loads.
__global__ void k_swizzle_W(const float* __restrict__ W, __bf16* __restrict__ Bsw, int ncol) {
    int idx = blockIdx.x * blockDim.x + threadIdx.x;
    if (idx >= 128 * ncol) return;
    int q    = idx & 15;
    int lane = (idx >> 4) & 31;
    int kt   = (idx >> 9) & 3;
    int ct   = idx >> 11;
    int K    = kt * 32 + ((lane >> 4) << 4) + q;
    int Nc   = ct * 16 + (lane & 15);
    Bsw[idx] = (__bf16)W[K * ncol + Nc];
}

// ---------------- WMMA GEMM: C[n,NCOL] = dinv[row] * (A[n,128] @ B[128,NCOL]) ----------------
// No LDS: each wave owns a 16-row strip; A fragments register-resident across all
// column tiles (2x b128 per kt), B fragments 2x b128 from the pre-swizzled L2-hot
// weights. K=128 via 4x wmma_f32_16x16x32_bf16. Branchless fast path; the lone
// tail block takes the guarded store path. OOB overreads (<=15 rows / <=60B dinv)
// stay inside workspace carve padding and only feed rows that are never stored.

template <int NCOL>
__global__ __launch_bounds__(256) void k_gemm_bf16(
    const __bf16* __restrict__ A,    // [n, 128] row-major bf16
    const __bf16* __restrict__ Bsw,  // swizzled [ (NCOL/16)*4*32*16 ]
    const float*  __restrict__ dinv, // [n]
    float* __restrict__ C,           // [n, NCOL]
    int n)
{
    const int wave = threadIdx.x >> 5;
    const int lane = threadIdx.x & 31;
    const int m16  = lane & 15;
    const int hi   = lane >> 4;
    const int rowStrip = (blockIdx.x * 8 + wave) * 16;
    if (rowStrip >= n) return;              // wave-uniform exit

    // A fragments for kt = 0..3 (16x32 bf16 each): per lane one contiguous 16-elem run (k = 16*hi + s).
    const __bf16* Arow = A + (size_t)(rowStrip + m16) * 128;
    union { v16bf v; bf16x8 h[2]; } ua[4];
#pragma unroll
    for (int kt = 0; kt < 4; ++kt) {
        ua[kt].h[0] = *(const bf16x8*)(Arow + kt * 32 + (hi << 4));
        ua[kt].h[1] = *(const bf16x8*)(Arow + kt * 32 + (hi << 4) + 8);
    }

    // ct-invariant: 8 dinv values for this lane's output rows (one 32B vector load).
    union { v8f v; float f[8]; } dv;
    dv.v = *(const v8f*)(dinv + rowStrip + (hi << 3));

    // Single store base; all store addresses are compile-time offsets from it.
    float* Cbase = C + (size_t)(rowStrip + (hi << 3)) * NCOL + m16;
    const bool fullStrip = (rowStrip + 16) <= n;

#pragma unroll
    for (int ct = 0; ct < NCOL / 16; ++ct) {
        union { v8f v; float f[8]; } acc;
#pragma unroll
        for (int j = 0; j < 8; ++j) acc.f[j] = 0.0f;

#pragma unroll
        for (int kt = 0; kt < 4; ++kt) {
            union { v16bf v; bf16x8 h[2]; } ub;
            const __bf16* bp = Bsw + (((ct * 4 + kt) * 32 + lane) << 4);
            ub.h[0] = *(const bf16x8*)(bp);
            ub.h[1] = *(const bf16x8*)(bp + 8);
            acc.v = __builtin_amdgcn_wmma_f32_16x16x32_bf16(
                false, ua[kt].v, false, ub.v, (short)0, acc.v, false, false);
        }

        // C/D layout: VGPR j -> M = j + 8*hi, N = m16
        if (fullStrip) {
#pragma unroll
            for (int j = 0; j < 8; ++j)
                Cbase[j * NCOL + ct * 16] = acc.f[j] * dv.f[j];
        } else {
#pragma unroll
            for (int j = 0; j < 8; ++j)
                if (rowStrip + (hi << 3) + j < n)
                    Cbase[j * NCOL + ct * 16] = acc.f[j] * dv.f[j];
        }
    }
}

// ---------------- edge scatter-add: agg[dst] += hs[src], one wave per edge ----------------

template <int NCOL>
__global__ __launch_bounds__(256) void k_edge_agg(
    const int* __restrict__ srcIdx, const int* __restrict__ dstIdx,
    const float* __restrict__ hs, float* __restrict__ agg, int E)
{
    constexpr int CH = NCOL / 32;   // floats per lane
    int e = blockIdx.x * 8 + (threadIdx.x >> 5);
    if (e >= E) return;
    // stream edge index lists ahead of the atomics
    __builtin_prefetch(srcIdx + e + 2048, 0, 0);
    __builtin_prefetch(dstIdx + e + 2048, 0, 0);
    int lane = threadIdx.x & 31;
    int s = srcIdx[e];
    int d = dstIdx[e];
    const float* hr = hs  + (size_t)s * NCOL + lane * CH;
    float*       ar = agg + (size_t)d * NCOL + lane * CH;
#pragma unroll
    for (int c = 0; c < CH; ++c) atomic_add_f32(&ar[c], hr[c]);
}

// ---------------- epilogues ----------------

// z = relu(dinv[i]*(agg + hs_selfloop) + b) -> bf16 (input to next GEMM)
__global__ void k_epilogue_relu_bf16(
    const float* __restrict__ agg, const float* __restrict__ hs,
    const float* __restrict__ dinv, const float* __restrict__ bias,
    __bf16* __restrict__ z, long long total)
{
    long long idx = (long long)blockIdx.x * blockDim.x + threadIdx.x;
    long long stride = (long long)gridDim.x * blockDim.x;
    for (; idx < total; idx += stride) {
        long long i = idx >> 7;          // / 128
        int c = (int)(idx & 127);
        float v = dinv[i] * (agg[idx] + hs[idx]) + bias[c];
        z[idx] = (__bf16)fmaxf(v, 0.0f);
    }
}

// out = dinv[i]*(agg + hs_selfloop) + b  (final fp32 output, width 64)
__global__ void k_epilogue_out(
    const float* __restrict__ agg, const float* __restrict__ hs,
    const float* __restrict__ dinv, const float* __restrict__ bias,
    float* __restrict__ out, long long total)
{
    long long idx = (long long)blockIdx.x * blockDim.x + threadIdx.x;
    long long stride = (long long)gridDim.x * blockDim.x;
    for (; idx < total; idx += stride) {
        long long i = idx >> 6;          // / 64
        int c = (int)(idx & 63);
        out[idx] = dinv[i] * (agg[idx] + hs[idx]) + bias[c];
    }
}

// ---------------- launch ----------------

extern "C" void kernel_launch(void* const* d_in, const int* in_sizes, int n_in,
                              void* d_out, int out_size, void* d_ws, size_t ws_size,
                              hipStream_t stream) {
    const float* x    = (const float*)d_in[0];
    const int*   edge = (const int*)d_in[1];
    const float* W1   = (const float*)d_in[2];
    const float* b1   = (const float*)d_in[3];
    const float* W2   = (const float*)d_in[4];
    const float* b2   = (const float*)d_in[5];

    const int n = in_sizes[0] / DIM_IN;
    const int E = in_sizes[1] / 2;
    const int* srcIdx = edge;
    const int* dstIdx = edge + E;

    // Workspace carve (with liveness-based reuse):
    char* ws = (char*)d_ws;
    size_t off = 0;
    auto carve = [&](size_t bytes) -> char* {
        char* p = ws + off;
        off += (bytes + 255) & ~(size_t)255;
        return p;
    };
    float*  dinv  = (float*)carve((size_t)n * 4 + 64);               // deg -> dinv (+tail pad)
    __bf16* W1sw  = (__bf16*)carve((size_t)DIM_IN * DIM_HID * 2);    // swizzled bf16 W1
    __bf16* W2sw  = (__bf16*)carve((size_t)DIM_HID * DIM_OUT * 2);   // swizzled bf16 W2
    __bf16* bufX  = (__bf16*)carve((size_t)n * 128 * 2 + 4096);      // x_bf16 / z1_bf16 (+pad)
    float*  hsBuf = (float*)carve((size_t)n * 128 * 4);              // hs1, then hs2
    float*  aggBuf= (float*)carve((size_t)n * 128 * 4);              // agg1, then agg2

    const int GS_BLOCKS = 2048;

    // 1) degrees -> dinv = rsqrt(deg + 1)
    k_zero_f32<<<GS_BLOCKS, 256, 0, stream>>>(dinv, (long long)n);
    k_deg_count<<<(E + 255) / 256, 256, 0, stream>>>(dstIdx, dinv, E);
    k_dinv<<<(n + 255) / 256, 256, 0, stream>>>(dinv, n);

    // 2) bf16 casts + weight swizzles (fragment-major for direct b128 fragment loads)
    k_cast_bf16<<<GS_BLOCKS, 256, 0, stream>>>(x, bufX, (long long)n * DIM_IN);
    k_swizzle_W<<<(128 * DIM_HID + 255) / 256, 256, 0, stream>>>(W1, W1sw, DIM_HID);
    k_swizzle_W<<<(128 * DIM_OUT + 255) / 256, 256, 0, stream>>>(W2, W2sw, DIM_OUT);

    const int gemmBlocks = (n + 127) / 128;   // 8 waves x 16 rows per block

    // 3) layer 1: hs1 = dinv * (x @ W1)
    k_gemm_bf16<DIM_HID><<<gemmBlocks, 256, 0, stream>>>(bufX, W1sw, dinv, hsBuf, n);

    // 4) aggregate layer 1 + fused epilogue (self-loop, scale, bias, relu, ->bf16)
    k_zero_f32<<<GS_BLOCKS, 256, 0, stream>>>(aggBuf, (long long)n * DIM_HID);
    k_edge_agg<DIM_HID><<<(E + 7) / 8, 256, 0, stream>>>(srcIdx, dstIdx, hsBuf, aggBuf, E);
    k_epilogue_relu_bf16<<<GS_BLOCKS, 256, 0, stream>>>(aggBuf, hsBuf, dinv, b1, bufX,
                                                        (long long)n * DIM_HID);

    // 5) layer 2: hs2 = dinv * (z1 @ W2)
    k_gemm_bf16<DIM_OUT><<<gemmBlocks, 256, 0, stream>>>(bufX, W2sw, dinv, hsBuf, n);

    // 6) aggregate layer 2 + final epilogue -> d_out (fp32)
    k_zero_f32<<<GS_BLOCKS, 256, 0, stream>>>(aggBuf, (long long)n * DIM_OUT);
    k_edge_agg<DIM_OUT><<<(E + 7) / 8, 256, 0, stream>>>(srcIdx, dstIdx, hsBuf, aggBuf, E);
    k_epilogue_out<<<GS_BLOCKS, 256, 0, stream>>>(aggBuf, hsBuf, dinv, b2, (float*)d_out,
                                                  (long long)n * DIM_OUT);
}
